// RotaryMultiHeadAttention_78683800863461
// MI455X (gfx1250) — matrix operand order repaired and checked by reading, not compiled, not run on hardware
//
#include <hip/hip_runtime.h>
#include <hip/hip_bf16.h>
#include <cstdint>

// Rotary MHA forward for gfx1250 (CDNA5, wave32, WMMA).
// Ping-pong double-buffered GEMMs (no register rotation copies),
// flash attention with online softmax. All matmuls: v_wmma_f32_16x16x32_f16.

typedef __attribute__((ext_vector_type(16))) _Float16 v16h;
typedef __attribute__((ext_vector_type(8)))  _Float16 v8h;
typedef __attribute__((ext_vector_type(8)))  float    v8f;
typedef _Float16 f16;

#define D_MODEL 1024
#define NHEADS  16
#define T_LEN   2048
#define D_K     64
#define NROWS   4096   // B * T_LEN

static __device__ __forceinline__ v8f wmma_f16(v16h a, v16h b, v8f c) {
  return __builtin_amdgcn_wmma_f32_16x16x32_f16(
      false, a, false, b, (short)0, c, false, false);
}

static __device__ __forceinline__ v16h load16h(const f16* p) {
  v8h lo = *(const v8h*)p;
  v8h hi = *(const v8h*)(p + 16);
  v16h r;
#pragma unroll
  for (int i = 0; i < 8; ++i) { r[i] = lo[i]; r[i + 8] = hi[i]; }
  return r;
}

static __device__ __forceinline__ v16h load16h_gap(const f16* p) {
  v8h lo = *(const v8h*)p;
  v8h hi = *(const v8h*)(p + 16);
  v16h r;
#pragma unroll
  for (int i = 0; i < 8; ++i) { r[i] = lo[i]; r[i + 8] = hi[i]; }
  return r;
}

// ---------------------------------------------------------------- convert
__global__ __launch_bounds__(256)
void cvt_f32_f16_kernel(const float* __restrict__ src, f16* __restrict__ dst, int n) {
  int i = blockIdx.x * blockDim.x + threadIdx.x;
  if (i < n) dst[i] = (f16)src[i];
}

// =====================================================================
// Shared ping-pong GEMM core: 32(M) x 64(N) per wave, K = D_MODEL in 32s.
// Two fragment sets alternate; loads write each set directly (no copies).
// Result left in acc[2][4].
// =====================================================================
#define GEMM_LOAD_SET(aset0, aset1, bset, Abase, Bbase, kk)                 \
  do {                                                                      \
    aset0 = load16h_gap(Abase + r0 * D_MODEL + (kk) + kbA);                 \
    aset1 = load16h_gap(Abase + r1 * D_MODEL + (kk) + kbA);                 \
    _Pragma("unroll")                                                       \
    for (int t = 0; t < 4; ++t)                                             \
      bset[t] = load16h(Bbase + (n0 + t * 16 + nn) * D_MODEL + (kk) + kbB); \
  } while (0)

#define GEMM_COMPUTE_SET(aset0, aset1, bset)                                \
  do {                                                                      \
    _Pragma("unroll")                                                       \
    for (int t = 0; t < 4; ++t) {                                           \
      acc[0][t] = wmma_f16(aset0, bset[t], acc[0][t]);                      \
      acc[1][t] = wmma_f16(aset1, bset[t], acc[1][t]);                      \
    }                                                                       \
  } while (0)

// ------------------------------------------------------- QKV projection
// mode 0: Q -> [B,H,T,DK]; mode 1: K -> [B,H,T,DK]; mode 2: V -> [B,H,DK,T]
__global__ __launch_bounds__(128)
void proj_qkv_kernel(const f16* __restrict__ xh, const f16* __restrict__ wh,
                     const float* __restrict__ bias, f16* __restrict__ out16,
                     int mode) {
  const int lane = threadIdx.x & 31;
  const int wid  = threadIdx.x >> 5;
  const int n0   = blockIdx.x * 64;
  const int m0   = blockIdx.y * 128 + wid * 32;
  const int r0   = m0 + (lane & 15);
  const int r1   = r0 + 16;
  const int kbA  = (lane >> 4) * 8;
  const int kbB  = (lane >> 4) * 8;
  const int nn   = lane & 15;

  v8f acc[2][4] = {};
  v16h aA0, aA1, bA[4];
  GEMM_LOAD_SET(aA0, aA1, bA, xh, wh, 0);

  int k0 = 0;
  for (; k0 < D_MODEL - 64; k0 += 64) {
    v16h aB0, aB1, bB[4];
    __builtin_prefetch(xh + r0 * D_MODEL + k0 + 256, 0, 1);
    GEMM_LOAD_SET(aB0, aB1, bB, xh, wh, k0 + 32);
    GEMM_COMPUTE_SET(aA0, aA1, bA);
    GEMM_LOAD_SET(aA0, aA1, bA, xh, wh, k0 + 64);
    GEMM_COMPUTE_SET(aB0, aB1, bB);
  }
  {
    v16h aB0, aB1, bB[4];
    GEMM_LOAD_SET(aB0, aB1, bB, xh, wh, k0 + 32);
    GEMM_COMPUTE_SET(aA0, aA1, bA);
    GEMM_COMPUTE_SET(aB0, aB1, bB);
  }

  // ---- bias + RoPE + store
#pragma unroll
  for (int t = 0; t < 4; ++t) {
    const int n  = n0 + t * 16 + nn;
    const int d  = n & (D_K - 1);
    const int h  = n >> 6;
    const float bn = bias[n];
    const float invf = __expf(-(float)((d >> 1) * 2) * (9.2103403719761836f / 64.0f));
#pragma unroll
    for (int mi = 0; mi < 2; ++mi) {
#pragma unroll
      for (int r = 0; r < 8; ++r) {
        const int grow = m0 + mi * 16 + (lane >> 4) * 8 + r;
        const int pos  = grow & (T_LEN - 1);
        const int bidx = grow >> 11;
        float val = acc[mi][t][r] + bn;
        if (mode < 2) {
          float ang = (float)pos * invf;
          float s = __sinf(ang), c = __cosf(ang);
          float other = __shfl_xor(val, 1, 32);
          val = ((d & 1) == 0) ? (val * c - other * s)
                               : (val * c + other * s);
          out16[(((size_t)bidx * NHEADS + h) * T_LEN + pos) * D_K + d] = (f16)val;
        } else {
          out16[(((size_t)bidx * NHEADS + h) * D_K + d) * T_LEN + pos] = (f16)val;
        }
      }
    }
  }
}

// --------------------------------------------------------- flash attention
__global__ __launch_bounds__(256)
void attn_kernel(const f16* __restrict__ Qh, const f16* __restrict__ Kh,
                 const f16* __restrict__ Vth, f16* __restrict__ ctxh) {
  __shared__ f16 ptile[8][16 * 32];

  const int lane = threadIdx.x & 31;
  const int wid  = threadIdx.x >> 5;
  const int bh   = blockIdx.y;
  const int b    = bh >> 4, h = bh & 15;
  const int q0   = blockIdx.x * 128 + wid * 16;

  const f16* Q  = Qh  + (size_t)bh * T_LEN * D_K;
  const f16* K  = Kh  + (size_t)bh * T_LEN * D_K;
  const f16* Vt = Vth + (size_t)bh * D_K * T_LEN;

  const int kbA = (lane >> 4) * 8;
  const int kbB = (lane >> 4) * 8;
  const int nn  = lane & 15;
  const int arow = q0 + nn;

  const v16h aq0 = load16h_gap(Q + arow * D_K + 0  + kbA);
  const v16h aq1 = load16h_gap(Q + arow * D_K + 32 + kbA);

  v8f o[4] = {};
  float mrow[8], lrow[8];
#pragma unroll
  for (int r = 0; r < 8; ++r) { mrow[r] = -1e30f; lrow[r] = 0.0f; }

  f16* pw = &ptile[wid][0];
  const int qhi = q0 + 15;                       // wave-uniform causal bound

  for (int j0 = 0; j0 <= qhi; j0 += 32) {
    // ---- K fragments first (consumed first by the score WMMAs)
    const int key0 = j0 + nn;
    const int key1 = key0 + 16;
    v16h bk00 = load16h(K + key0 * D_K + 0  + kbB);
    v16h bk01 = load16h(K + key0 * D_K + 32 + kbB);
    v16h bk10 = load16h(K + key1 * D_K + 0  + kbB);
    v16h bk11 = load16h(K + key1 * D_K + 32 + kbB);
    // ---- V fragments: consumed last, latency hides behind WMMAs + softmax
    v16h bv[4];
#pragma unroll
    for (int t = 0; t < 4; ++t)
      bv[t] = load16h(Vt + (t * 16 + nn) * T_LEN + j0 + kbB);

    v8f s0 = {}, s1 = {};
    s0 = wmma_f16(aq0, bk00, s0);
    s0 = wmma_f16(aq1, bk01, s0);
    s1 = wmma_f16(aq0, bk10, s1);
    s1 = wmma_f16(aq1, bk11, s1);

    // ---- online softmax (row stats across each 16-lane group)
#pragma unroll
    for (int r = 0; r < 8; ++r) {
      const int qm = q0 + (lane >> 4) * 8 + r;
      float x0 = (key0 <= qm) ? s0[r] * 0.125f : -1e30f;   // 1/sqrt(64)
      float x1 = (key1 <= qm) ? s1[r] * 0.125f : -1e30f;
      float mx = fmaxf(x0, x1);
#pragma unroll
      for (int off = 1; off < 16; off <<= 1)
        mx = fmaxf(mx, __shfl_xor(mx, off, 32));
      const float mnew  = fmaxf(mrow[r], mx);
      const float scale = __expf(mrow[r] - mnew);
      const float p0 = __expf(x0 - mnew);
      const float p1 = __expf(x1 - mnew);
      float rs = p0 + p1;
#pragma unroll
      for (int off = 1; off < 16; off <<= 1)
        rs += __shfl_xor(rs, off, 32);
      lrow[r] = lrow[r] * scale + rs;
      mrow[r] = mnew;
#pragma unroll
      for (int t = 0; t < 4; ++t) o[t][r] *= scale;
      const int m = (lane >> 4) * 8 + r;
      pw[m * 32 + nn]      = (f16)p0;
      pw[m * 32 + 16 + nn] = (f16)p1;
    }

    // per-wave LDS RAW: LDS is in-order within a wave; fence the compiler.
    asm volatile("s_wait_dscnt 0" ::: "memory");

    // ---- O += P (16x32) x V (32x64)
    const v16h pa = load16h_gap(pw + nn * 32 + kbA);
#pragma unroll
    for (int t = 0; t < 4; ++t)
      o[t] = wmma_f16(pa, bv[t], o[t]);
  }

  // ---- normalize and store context f16 [B*T, D_MODEL]
#pragma unroll
  for (int r = 0; r < 8; ++r) {
    const float inv = 1.0f / lrow[r];
    const int qm = q0 + (lane >> 4) * 8 + r;
#pragma unroll
    for (int t = 0; t < 4; ++t) {
      const int d = t * 16 + nn;
      ctxh[((size_t)(b * T_LEN + qm)) * D_MODEL + h * D_K + d] = (f16)(o[t][r] * inv);
    }
  }
}

// ------------------------------------------------------ output projection
__global__ __launch_bounds__(128)
void proj_out_kernel(const f16* __restrict__ ctxh, const f16* __restrict__ woh,
                     const float* __restrict__ bo, float* __restrict__ out) {
  const int lane = threadIdx.x & 31;
  const int wid  = threadIdx.x >> 5;
  const int n0   = blockIdx.x * 64;
  const int m0   = blockIdx.y * 128 + wid * 32;
  const int r0   = m0 + (lane & 15);
  const int r1   = r0 + 16;
  const int kbA  = (lane >> 4) * 8;
  const int kbB  = (lane >> 4) * 8;
  const int nn   = lane & 15;

  v8f acc[2][4] = {};
  v16h aA0, aA1, bA[4];
  GEMM_LOAD_SET(aA0, aA1, bA, ctxh, woh, 0);

  int k0 = 0;
  for (; k0 < D_MODEL - 64; k0 += 64) {
    v16h aB0, aB1, bB[4];
    __builtin_prefetch(ctxh + r0 * D_MODEL + k0 + 256, 0, 1);
    GEMM_LOAD_SET(aB0, aB1, bB, ctxh, woh, k0 + 32);
    GEMM_COMPUTE_SET(aA0, aA1, bA);
    GEMM_LOAD_SET(aA0, aA1, bA, ctxh, woh, k0 + 64);
    GEMM_COMPUTE_SET(aB0, aB1, bB);
  }
  {
    v16h aB0, aB1, bB[4];
    GEMM_LOAD_SET(aB0, aB1, bB, ctxh, woh, k0 + 32);
    GEMM_COMPUTE_SET(aA0, aA1, bA);
    GEMM_COMPUTE_SET(aB0, aB1, bB);
  }

#pragma unroll
  for (int t = 0; t < 4; ++t) {
    const int n = n0 + t * 16 + nn;
    const float bn = bo[n];
#pragma unroll
    for (int mi = 0; mi < 2; ++mi) {
#pragma unroll
      for (int r = 0; r < 8; ++r) {
        const int grow = m0 + mi * 16 + (lane >> 4) * 8 + r;
        out[(size_t)grow * D_MODEL + n] = acc[mi][t][r] + bn;
      }
    }
  }
}

// ---------------------------------------------------------------- launch
extern "C" void kernel_launch(void* const* d_in, const int* in_sizes, int n_in,
                              void* d_out, int out_size, void* d_ws, size_t ws_size,
                              hipStream_t stream) {
  const float* x  = (const float*)d_in[0];
  const float* wq = (const float*)d_in[1];
  const float* bq = (const float*)d_in[2];
  const float* wk = (const float*)d_in[3];
  const float* bk = (const float*)d_in[4];
  const float* wv = (const float*)d_in[5];
  const float* bv = (const float*)d_in[6];
  const float* wo = (const float*)d_in[7];
  const float* bo = (const float*)d_in[8];
  float* out = (float*)d_out;

  char* ws = (char*)d_ws;
  size_t off = 0;
  f16* xh   = (f16*)(ws + off); off += (size_t)NROWS * D_MODEL * 2;
  f16* wqh  = (f16*)(ws + off); off += (size_t)D_MODEL * D_MODEL * 2;
  f16* wkh  = (f16*)(ws + off); off += (size_t)D_MODEL * D_MODEL * 2;
  f16* wvh  = (f16*)(ws + off); off += (size_t)D_MODEL * D_MODEL * 2;
  f16* woh  = (f16*)(ws + off); off += (size_t)D_MODEL * D_MODEL * 2;
  f16* Qh   = (f16*)(ws + off); off += (size_t)NROWS * D_MODEL * 2;
  f16* Khh  = (f16*)(ws + off); off += (size_t)NROWS * D_MODEL * 2;
  f16* Vth  = (f16*)(ws + off); off += (size_t)NROWS * D_MODEL * 2;
  f16* ctxh = (f16*)(ws + off); off += (size_t)NROWS * D_MODEL * 2;

  const int nx = NROWS * D_MODEL;
  const int nw = D_MODEL * D_MODEL;
  cvt_f32_f16_kernel<<<(nx + 255) / 256, 256, 0, stream>>>(x,  xh,  nx);
  cvt_f32_f16_kernel<<<(nw + 255) / 256, 256, 0, stream>>>(wq, wqh, nw);
  cvt_f32_f16_kernel<<<(nw + 255) / 256, 256, 0, stream>>>(wk, wkh, nw);
  cvt_f32_f16_kernel<<<(nw + 255) / 256, 256, 0, stream>>>(wv, wvh, nw);
  cvt_f32_f16_kernel<<<(nw + 255) / 256, 256, 0, stream>>>(wo, woh, nw);

  dim3 gProj(D_MODEL / 64, NROWS / 128);   // (16, 32)
  proj_qkv_kernel<<<gProj, 128, 0, stream>>>(xh, wqh, bq, Qh,  0);
  proj_qkv_kernel<<<gProj, 128, 0, stream>>>(xh, wkh, bk, Khh, 1);
  proj_qkv_kernel<<<gProj, 128, 0, stream>>>(xh, wvh, bv, Vth, 2);

  dim3 gAttn(T_LEN / 128, 2 * NHEADS);     // (16, 32)
  attn_kernel<<<gAttn, 256, 0, stream>>>(Qh, Khh, Vth, ctxh);

  proj_out_kernel<<<gProj, 128, 0, stream>>>(ctxh, woh, bo, out);
}
